// NATLayer_72679436583286
// MI455X (gfx1250) — matrix operand order repaired and checked by reading, not compiled, not run on hardware
//
#include <hip/hip_runtime.h>
#include <cstddef>
#include <cstdint>

// ---------------------------------------------------------------------------
// NAT transformer layer for MI455X (gfx1250, wave32, WMMA).
// GEMMs: v_wmma_f32_16x16x32_bf16, 128x128 block tile, 32x64 per wave,
// double-buffered LDS fed by GLOBAL_LOAD_ASYNC_TO_LDS_B128 (ASYNCcnt),
// weights pre-transposed to N x K so all fragment reads are ds_load_b128.
// ---------------------------------------------------------------------------

typedef __attribute__((ext_vector_type(16))) __bf16 v16bf;
typedef __attribute__((ext_vector_type(8)))  __bf16 v8bf;
typedef __attribute__((ext_vector_type(8)))  float  v8f;

#define BDIM   2
#define LDIM   2048
#define DMODEL 1024
#define NHEAD  16
#define HDIM   64
#define RAD    7
#define WINW   15
#define HIDDEN 4096
#define MROWS  (BDIM * LDIM)          // 4096
#define QKVN   (3 * DMODEL)           // 3072

__device__ __forceinline__ unsigned short f2bf(float f) {
    unsigned int u = __builtin_bit_cast(unsigned int, f);
    u = (u + 0x7FFFu + ((u >> 16) & 1u)) >> 16;   // round-to-nearest-even
    return (unsigned short)u;
}

// ---- CDNA5 async global->LDS copy: 2 x 16B per lane via INST_OFFSET -------
// ISA: the instruction offset is added to both the LDS and the memory address -> the same
// offset advances BOTH sides, so consecutive 16B chunks reuse the same VGPRs.
__device__ __forceinline__ void async_copy_b256(unsigned lds_addr, const void* gaddr) {
    asm volatile("global_load_async_to_lds_b128 %0, %1, off\n\t"
                 "global_load_async_to_lds_b128 %0, %1, off offset:16"
                 :: "v"(lds_addr), "v"(gaddr) : "memory");
}
__device__ __forceinline__ void wait_async0() {
    asm volatile("s_wait_asynccnt 0x0" ::: "memory");
}
// generic->LDS address: low 32 bits of a __shared__ generic pointer
__device__ __forceinline__ unsigned lds_addr_of(const void* p) {
    return (unsigned)(uintptr_t)p;
}

// ---------------------------------------------------------------------------
// Transpose + convert: in[K x N] f32 -> out[N x K] bf16. 32x32 LDS tiles,
// coalesced on both sides. One-time cost per weight matrix.
// ---------------------------------------------------------------------------
__global__ __launch_bounds__(256) void k_convT_bf16(
    const float* __restrict__ in, unsigned short* __restrict__ out, int K, int N)
{
    __shared__ unsigned short tile[32][33];
    const int nb = blockIdx.x * 32;
    const int kb = blockIdx.y * 32;
    const int tx = threadIdx.x & 31;
    const int ty = threadIdx.x >> 5;       // 0..7
    #pragma unroll
    for (int i = 0; i < 4; ++i) {
        const int k = kb + ty + i * 8;
        tile[ty + i * 8][tx] = f2bf(in[(size_t)k * N + nb + tx]);
    }
    __syncthreads();
    #pragma unroll
    for (int i = 0; i < 4; ++i) {
        const int n = nb + ty + i * 8;
        out[(size_t)n * K + kb + tx] = tile[tx][ty + i * 8];
    }
}

// ---------------------------------------------------------------------------
// LayerNorm over rows of DMODEL, bf16 output. One block (256 thr) per row.
// ---------------------------------------------------------------------------
__global__ __launch_bounds__(256) void k_layernorm_bf16(
    const float* __restrict__ x, const float* __restrict__ g,
    const float* __restrict__ b, unsigned short* __restrict__ out)
{
    __shared__ float s1[256];
    __shared__ float s2[256];
    const int row = blockIdx.x;
    const int t   = threadIdx.x;

    float4 v = reinterpret_cast<const float4*>(x + (size_t)row * DMODEL)[t];
    float s = v.x + v.y + v.z + v.w;
    float q = v.x * v.x + v.y * v.y + v.z * v.z + v.w * v.w;
    s1[t] = s; s2[t] = q;
    __syncthreads();
    #pragma unroll
    for (int o = 128; o > 0; o >>= 1) {
        if (t < o) { s1[t] += s1[t + o]; s2[t] += s2[t + o]; }
        __syncthreads();
    }
    const float mu   = s1[0] * (1.0f / DMODEL);
    const float var  = s2[0] * (1.0f / DMODEL) - mu * mu;
    const float rstd = rsqrtf(var + 1e-5f);

    float4 gv = reinterpret_cast<const float4*>(g)[t];
    float4 bv = reinterpret_cast<const float4*>(b)[t];
    unsigned short* op = out + (size_t)row * DMODEL + t * 4;
    op[0] = f2bf((v.x - mu) * rstd * gv.x + bv.x);
    op[1] = f2bf((v.y - mu) * rstd * gv.y + bv.y);
    op[2] = f2bf((v.z - mu) * rstd * gv.z + bv.z);
    op[3] = f2bf((v.w - mu) * rstd * gv.w + bv.w);
}

// ---------------------------------------------------------------------------
// bf16 WMMA GEMM:  out[M,N] = A[M,K](bf16) * Bt[N,K](bf16)^T + bias [+res][GELU]
//
// Block: 256 threads = 8 waves; tile 128(M) x 128(N), BK = 32.
// Waves: 4(M) x 2(N); each wave computes 32x64 = 8 WMMA accumulators.
// Double-buffered LDS filled by async global->LDS b128 copies; one
// s_wait_asynccnt + barrier per K-step, copies overlap the 8 WMMAs.
// Fragment layouts (CDNA5 16-bit WMMA):
//   A: lane(l) row = lm, K = lh*8 + {0..7} and {16..23}
//   B: lane(l) col = lm, K = lh*8 + {0..7} and {16..23}
// ---------------------------------------------------------------------------
template<bool GELU, bool HAS_RES, bool OUT_BF16>
__global__ __launch_bounds__(256) void k_gemm_bf16(
    const unsigned short* __restrict__ A,   // M x K (row-major)
    const unsigned short* __restrict__ Bt,  // N x K (row-major, pre-transposed)
    const float* __restrict__ bias, const float* __restrict__ res,
    void* __restrict__ outp, int M, int N, int K)
{
    constexpr int LDT = 40;   // LDS row stride (elems): 32 + 8 pad
    __shared__ __align__(16) unsigned short As[2][128 * LDT];
    __shared__ __align__(16) unsigned short Bs[2][128 * LDT];

    const int tid  = threadIdx.x;
    const int lane = tid & 31;
    const int wid  = tid >> 5;
    const int wm   = wid & 3;            // m-wave: 32 rows each
    const int wn   = wid >> 2;           // n-wave: 64 cols each
    const int m0   = blockIdx.y * 128;
    const int n0   = blockIdx.x * 128;

    // staging: thread -> (row = tid>>1 of 128, 16 contiguous k at (tid&1)*16)
    const int g_r   = tid >> 1;
    const int g_c   = (tid & 1) << 4;
    const int stoff = g_r * LDT + g_c;
    const unsigned short* gA = A  + (size_t)(m0 + g_r) * K + g_c;
    const unsigned short* gB = Bt + (size_t)(n0 + g_r) * K + g_c;

    // per-thread LDS destinations (byte addresses) for async copies
    const unsigned ldsA[2] = { lds_addr_of(&As[0][stoff]), lds_addr_of(&As[1][stoff]) };
    const unsigned ldsB[2] = { lds_addr_of(&Bs[0][stoff]), lds_addr_of(&Bs[1][stoff]) };

    const int lh = lane >> 4;
    const int lm = lane & 15;
    const int aoff = (wm * 32 + lm) * LDT + lh * 8;    // + s*16*LDT
    const int boff = (wn * 64 + lm) * LDT + lh * 8;    // + t*16*LDT

    v8f acc[2][4];
    #pragma unroll
    for (int s = 0; s < 2; ++s)
        #pragma unroll
        for (int t = 0; t < 4; ++t)
            acc[s][t] = {};

    auto stage = [&](int buf, int kt) {
        async_copy_b256(ldsA[buf], gA + (size_t)kt * 32);
        async_copy_b256(ldsB[buf], gB + (size_t)kt * 32);
    };

    auto mma_step = [&](int cur) {
        const unsigned short* as = &As[cur][0];
        const unsigned short* bs = &Bs[cur][0];
        v16bf af[2];
        #pragma unroll
        for (int s = 0; s < 2; ++s) {
            const unsigned short* p = as + aoff + s * (16 * LDT);
            v8bf lo = *reinterpret_cast<const v8bf*>(p);
            v8bf hi = *reinterpret_cast<const v8bf*>(p + 16);
            af[s] = __builtin_shufflevector(lo, hi,
                0,1,2,3,4,5,6,7,8,9,10,11,12,13,14,15);
        }
        #pragma unroll
        for (int t = 0; t < 4; ++t) {
            const unsigned short* p = bs + boff + t * (16 * LDT);
            v8bf lo = *reinterpret_cast<const v8bf*>(p);
            v8bf hi = *reinterpret_cast<const v8bf*>(p + 16);
            v16bf bfr = __builtin_shufflevector(lo, hi,
                0,1,2,3,4,5,6,7,8,9,10,11,12,13,14,15);
            #pragma unroll
            for (int s = 0; s < 2; ++s)
                acc[s][t] = __builtin_amdgcn_wmma_f32_16x16x32_bf16(
                    false, af[s], false, bfr, (short)0, acc[s][t], false, false);
        }
    };

    const int nk = K >> 5;

    // prologue: stage tile 0 into buf0
    stage(0, 0);
    wait_async0();
    __syncthreads();

    for (int kt = 0; kt < nk - 1; ++kt) {
        const int cur = kt & 1;
        const int nxt = cur ^ 1;
        // issue async copies for next tile; they run while we do 8 WMMAs.
        // buf[nxt] was last READ in iter kt-1 and that iteration's barrier
        // ordered those reads before these writes.
        stage(nxt, kt + 1);
        mma_step(cur);
        wait_async0();
        __syncthreads();
    }
    mma_step((nk - 1) & 1);

    // ---- epilogue: C/D layout lane -> (m = r + 8*lh, n = lm)
    #pragma unroll
    for (int s = 0; s < 2; ++s) {
        #pragma unroll
        for (int t = 0; t < 4; ++t) {
            const int n = n0 + wn * 64 + t * 16 + lm;
            const float bn = bias[n];
            #pragma unroll
            for (int r = 0; r < 8; ++r) {
                const int m = m0 + wm * 32 + s * 16 + r + 8 * lh;
                float v = acc[s][t][r] + bn;
                if (HAS_RES) v += res[(size_t)m * N + n];
                if (GELU)    v = 0.5f * v * (1.0f + erff(v * 0.70710678118654752f));
                if (OUT_BF16)
                    reinterpret_cast<unsigned short*>(outp)[(size_t)m * N + n] = f2bf(v);
                else
                    reinterpret_cast<float*>(outp)[(size_t)m * N + n] = v;
            }
        }
    }
}

// ---------------------------------------------------------------------------
// Neighborhood attention: one wave per (b, l, head). Lane owns dims d, d+32.
// Matches reference exactly: zero-padded K gives score 0 for OOB positions,
// which still participate in the softmax denominator; V OOB contributes 0.
// ---------------------------------------------------------------------------
__global__ __launch_bounds__(256) void k_nat_attn(
    const float* __restrict__ qkv, unsigned short* __restrict__ out)
{
    const int lane = threadIdx.x & 31;
    const int widx = blockIdx.x * 8 + (threadIdx.x >> 5);  // 0 .. B*L*NHEAD-1
    const int h    = widx & (NHEAD - 1);
    const int tmp  = widx >> 4;
    const int l    = tmp & (LDIM - 1);
    const int b    = tmp >> 11;                            // tmp / LDIM

    const float* qp = qkv + (size_t)(b * LDIM + l) * QKVN + h * HDIM;
    const float q0 = qp[lane];
    const float q1 = qp[lane + 32];

    float sc[WINW];
    #pragma unroll
    for (int j = 0; j < WINW; ++j) {
        const int p = l + j - RAD;
        const bool in = (p >= 0) && (p < LDIM);
        float s = 0.0f;
        if (in) {
            const float* kp = qkv + (size_t)(b * LDIM + p) * QKVN + DMODEL + h * HDIM;
            s = q0 * kp[lane] + q1 * kp[lane + 32];
        }
        #pragma unroll
        for (int o = 16; o > 0; o >>= 1) s += __shfl_xor(s, o, 32);
        sc[j] = in ? s * 0.125f : 0.0f;                    // /sqrt(64)
    }

    float mx = sc[0];
    #pragma unroll
    for (int j = 1; j < WINW; ++j) mx = fmaxf(mx, sc[j]);
    float w[WINW];
    float den = 0.0f;
    #pragma unroll
    for (int j = 0; j < WINW; ++j) { w[j] = expf(sc[j] - mx); den += w[j]; }
    const float inv = 1.0f / den;

    float o0 = 0.0f, o1 = 0.0f;
    #pragma unroll
    for (int j = 0; j < WINW; ++j) {
        const int p = l + j - RAD;
        if (p >= 0 && p < LDIM) {
            const float* vp = qkv + (size_t)(b * LDIM + p) * QKVN + 2 * DMODEL + h * HDIM;
            const float wj = w[j] * inv;
            o0 += wj * vp[lane];
            o1 += wj * vp[lane + 32];
        }
    }

    unsigned short* op = out + (size_t)(b * LDIM + l) * DMODEL + h * HDIM;
    op[lane]      = f2bf(o0);
    op[lane + 32] = f2bf(o1);
}

// ---------------------------------------------------------------------------
// Host-side orchestration
// ---------------------------------------------------------------------------
extern "C" void kernel_launch(void* const* d_in, const int* in_sizes, int n_in,
                              void* d_out, int out_size, void* d_ws, size_t ws_size,
                              hipStream_t stream)
{
    (void)in_sizes; (void)n_in; (void)out_size; (void)ws_size;

    const float* x      = (const float*)d_in[0];
    const float* qkv_w  = (const float*)d_in[1];
    const float* qkv_b  = (const float*)d_in[2];
    const float* proj_w = (const float*)d_in[3];
    const float* proj_b = (const float*)d_in[4];
    const float* n1g    = (const float*)d_in[5];
    const float* n1b    = (const float*)d_in[6];
    const float* n2g    = (const float*)d_in[7];
    const float* n2b    = (const float*)d_in[8];
    const float* fc1_w  = (const float*)d_in[9];
    const float* fc1_b  = (const float*)d_in[10];
    const float* fc2_w  = (const float*)d_in[11];
    const float* fc2_b  = (const float*)d_in[12];
    float* out = (float*)d_out;

    // workspace carve-up
    char* ws = (char*)d_ws;
    size_t off = 0;
    auto carve = [&](size_t bytes) -> void* {
        void* p = ws + off;
        off += (bytes + 255) & ~(size_t)255;
        return p;
    };
    unsigned short* wqkv_t = (unsigned short*)carve((size_t)DMODEL * QKVN   * 2); // [3072 x 1024]
    unsigned short* wprj_t = (unsigned short*)carve((size_t)DMODEL * DMODEL * 2); // [1024 x 1024]
    unsigned short* wfc1_t = (unsigned short*)carve((size_t)DMODEL * HIDDEN * 2); // [4096 x 1024]
    unsigned short* wfc2_t = (unsigned short*)carve((size_t)HIDDEN * DMODEL * 2); // [1024 x 4096]
    unsigned short* ln_bf  = (unsigned short*)carve((size_t)MROWS  * DMODEL * 2);
    float*          qkv_f  = (float*)         carve((size_t)MROWS  * QKVN   * 4);
    unsigned short* attn_bf= (unsigned short*)carve((size_t)MROWS  * DMODEL * 2);
    float*          x1_f   = (float*)         carve((size_t)MROWS  * DMODEL * 4);
    unsigned short* h_bf   = (unsigned short*)carve((size_t)MROWS  * HIDDEN * 2);

    // 1) weights -> bf16, transposed to N x K
    k_convT_bf16<<<dim3(QKVN   / 32, DMODEL / 32), 256, 0, stream>>>(qkv_w,  wqkv_t, DMODEL, QKVN);
    k_convT_bf16<<<dim3(DMODEL / 32, DMODEL / 32), 256, 0, stream>>>(proj_w, wprj_t, DMODEL, DMODEL);
    k_convT_bf16<<<dim3(HIDDEN / 32, DMODEL / 32), 256, 0, stream>>>(fc1_w,  wfc1_t, DMODEL, HIDDEN);
    k_convT_bf16<<<dim3(DMODEL / 32, HIDDEN / 32), 256, 0, stream>>>(fc2_w,  wfc2_t, HIDDEN, DMODEL);

    // 2) LN1 -> bf16
    k_layernorm_bf16<<<MROWS, 256, 0, stream>>>(x, n1g, n1b, ln_bf);

    // 3) QKV GEMM: (4096 x 1024) x (1024 x 3072) + bias -> f32
    k_gemm_bf16<false, false, false>
        <<<dim3(QKVN / 128, MROWS / 128), 256, 0, stream>>>(
            ln_bf, wqkv_t, qkv_b, nullptr, qkv_f, MROWS, QKVN, DMODEL);

    // 4) neighborhood attention -> bf16
    k_nat_attn<<<(MROWS * NHEAD) / 8, 256, 0, stream>>>(qkv_f, attn_bf);

    // 5) proj GEMM + residual(x) -> x1 (f32)
    k_gemm_bf16<false, true, false>
        <<<dim3(DMODEL / 128, MROWS / 128), 256, 0, stream>>>(
            attn_bf, wprj_t, proj_b, x, x1_f, MROWS, DMODEL, DMODEL);

    // 6) LN2 -> bf16
    k_layernorm_bf16<<<MROWS, 256, 0, stream>>>(x1_f, n2g, n2b, ln_bf);

    // 7) FC1 GEMM + bias + exact GELU -> bf16
    k_gemm_bf16<true, false, true>
        <<<dim3(HIDDEN / 128, MROWS / 128), 256, 0, stream>>>(
            ln_bf, wfc1_t, fc1_b, nullptr, h_bf, MROWS, HIDDEN, DMODEL);

    // 8) FC2 GEMM + bias + residual(x1) -> d_out (f32)
    k_gemm_bf16<false, true, false>
        <<<dim3(DMODEL / 128, MROWS / 128), 256, 0, stream>>>(
            h_bf, wfc2_t, fc2_b, x1_f, out, MROWS, DMODEL, HIDDEN);
}
